// GraphQLayer_65481071399741
// MI455X (gfx1250) — compile-verified, matrix operand order checked by reading, not run on hardware
//
#include <hip/hip_runtime.h>

#define NN 16384
#define DF 128
#define HID 64
#define THRESH 0.85f

typedef __attribute__((ext_vector_type(16))) __bf16 bf16x16;
typedef __attribute__((ext_vector_type(8)))  __bf16 bf16x8;
typedef __attribute__((ext_vector_type(8)))  float  f32x8;

// Load a 16x32 bf16 WMMA A/B fragment slice for this lane from a row-major
// 128-feature row. Per ISA layout: lanes 0-15 hold K = k0+0..7 and k0+16..23,
// lanes 16-31 hold K = k0+8..15 and k0+24..31 (packed 2 per VGPR). Both A and
// B fragments reduce to the same two 16-byte contiguous loads.
__device__ __forceinline__ bf16x16 load_frag(const __bf16* row, int k0, int half) {
    union { bf16x16 v; bf16x8 h[2]; } u;
    u.h[0] = *reinterpret_cast<const bf16x8*>(row + k0 + half * 8);
    u.h[1] = *reinterpret_cast<const bf16x8*>(row + k0 + 16 + half * 8);
    return u.v;
}

// ---- prep: split x into bf16 hi/lo parts (elementwise) ----
__global__ __launch_bounds__(256) void split_kernel(const float* __restrict__ x,
                                                    __bf16* __restrict__ xh,
                                                    __bf16* __restrict__ xl) {
    int g = blockIdx.x * 256 + threadIdx.x;
    float v = x[g];
    __bf16 h = (__bf16)v;
    __bf16 l = (__bf16)(v - (float)h);
    xh[g] = h;
    xl[g] = l;
}

// ---- prep: per-node feature mean + zero the scalar accumulator ----
__global__ __launch_bounds__(256) void rowmean_kernel(const float* __restrict__ x,
                                                      float* __restrict__ rowmean,
                                                      float* __restrict__ s) {
    int wave = threadIdx.x >> 5;
    int lane = threadIdx.x & 31;
    int row  = blockIdx.x * 8 + wave;
    const float* p = x + row * DF;
    float sum = p[lane] + p[lane + 32] + p[lane + 64] + p[lane + 96];
    sum += __shfl_xor(sum, 1,  32);
    sum += __shfl_xor(sum, 2,  32);
    sum += __shfl_xor(sum, 4,  32);
    sum += __shfl_xor(sum, 8,  32);
    sum += __shfl_xor(sum, 16, 32);
    if (lane == 0) {
        rowmean[row] = sum * (1.0f / DF);
        s[row] = 0.0f;
    }
}

// ---- prep: row sums of W ----
__global__ __launch_bounds__(64) void wsum_kernel(const float* __restrict__ W,
                                                  float* __restrict__ wsum) {
    int h = threadIdx.x;
    float acc = 0.0f;
    #pragma unroll 4
    for (int d = 0; d < DF; ++d) acc += W[h * DF + d];
    wsum[h] = acc;
}

// ---- main: fused Gram(x) -> threshold -> scalar neighbor-mean accumulation ----
// Grid: NN/64 blocks of 256 threads (8 waves = 2 M-groups x 4 N-tiles).
// Each wave computes TWO 16-row subtiles against the same streamed B fragments
// (24 WMMAs per 16 b128 loads -> 48 FLOP/B from L1 instead of 24).
__global__ __launch_bounds__(256) void pair_kernel(const __bf16* __restrict__ xh,
                                                   const __bf16* __restrict__ xl,
                                                   const float*  __restrict__ rowmean,
                                                   float*        __restrict__ s) {
    const int lane   = threadIdx.x & 31;
    const int wave   = threadIdx.x >> 5;
    const int wave_m = wave >> 2;   // 0..1
    const int wave_n = wave & 3;    // 0..3
    const int half   = lane >> 4;   // 0..1
    const int idx    = lane & 15;
    const int m0     = blockIdx.x * 64 + wave_m * 32;  // this wave's 32 rows

    // Register-resident A fragments: 2 subtiles x (hi,lo) x 4 k-steps
    bf16x16 Ah[2][4], Al[2][4];
    #pragma unroll
    for (int t = 0; t < 2; ++t) {
        const __bf16* arow_h = xh + (m0 + t * 16 + idx) * DF;
        const __bf16* arow_l = xl + (m0 + t * 16 + idx) * DF;
        #pragma unroll
        for (int k = 0; k < 4; ++k) {
            Ah[t][k] = load_frag(arow_h, k * 32, half);
            Al[t][k] = load_frag(arow_l, k * 32, half);
        }
    }

    // Per-lane scalar accumulators for this lane's 16 fixed output rows
    float acc_s[2][8];
    #pragma unroll
    for (int t = 0; t < 2; ++t)
        #pragma unroll
        for (int r = 0; r < 8; ++r) acc_s[t][r] = 0.0f;

    for (int jb = 0; jb < NN; jb += 64) {
        const int col = jb + wave_n * 16 + idx;   // this lane's B column (node j)
        const __bf16* brow_h = xh + col * DF;
        const __bf16* brow_l = xl + col * DF;

        // 6 independent accumulator chains: [subtile][hh,hl,lh]
        f32x8 c[2][3] = {{{}, {}, {}}, {{}, {}, {}}};
        #pragma unroll
        for (int k = 0; k < 4; ++k) {
            bf16x16 Bh = load_frag(brow_h, k * 32, half);
            bf16x16 Bl = load_frag(brow_l, k * 32, half);
            #pragma unroll
            for (int t = 0; t < 2; ++t) {
                c[t][0] = __builtin_amdgcn_wmma_f32_16x16x32_bf16(
                              false, Ah[t][k], false, Bh, (short)0, c[t][0], false, false);
                c[t][1] = __builtin_amdgcn_wmma_f32_16x16x32_bf16(
                              false, Ah[t][k], false, Bl, (short)0, c[t][1], false, false);
                c[t][2] = __builtin_amdgcn_wmma_f32_16x16x32_bf16(
                              false, Al[t][k], false, Bh, (short)0, c[t][2], false, false);
            }
        }

        const float rm = rowmean[col];
        #pragma unroll
        for (int t = 0; t < 2; ++t) {
            #pragma unroll
            for (int r = 0; r < 8; ++r) {
                float d = c[t][0][r] + c[t][1][r] + c[t][2][r];  // ~fp32-accurate dot
                int row = m0 + t * 16 + half * 8 + r;
                bool edge = (d * d >= THRESH) && (row != col);
                acc_s[t][r] += edge ? rm : 0.0f;
            }
        }
    }

    // Reduce the 16 lanes sharing each row (xor masks < 16 stay in-half),
    // then one atomic per row to combine the 4 wave_n waves.
    #pragma unroll
    for (int t = 0; t < 2; ++t) {
        #pragma unroll
        for (int r = 0; r < 8; ++r) {
            float v = acc_s[t][r];
            v += __shfl_xor(v, 1, 32);
            v += __shfl_xor(v, 2, 32);
            v += __shfl_xor(v, 4, 32);
            v += __shfl_xor(v, 8, 32);
            if (idx == 0) atomicAdd(&s[m0 + t * 16 + half * 8 + r], v);
        }
    }
}

// ---- final: out[i,h] = s[i] * wsum[h] + b[h] ----
__global__ __launch_bounds__(256) void out_kernel(const float* __restrict__ s,
                                                  const float* __restrict__ wsum,
                                                  const float* __restrict__ b,
                                                  float* __restrict__ out) {
    int g = blockIdx.x * 256 + threadIdx.x;
    int i = g >> 6;
    int h = g & 63;
    out[g] = s[i] * wsum[h] + b[h];
}

extern "C" void kernel_launch(void* const* d_in, const int* in_sizes, int n_in,
                              void* d_out, int out_size, void* d_ws, size_t ws_size,
                              hipStream_t stream) {
    const float* x = (const float*)d_in[0];   // [16384, 128]
    const float* W = (const float*)d_in[1];   // [64, 128]
    const float* b = (const float*)d_in[2];   // [64]
    float* out = (float*)d_out;               // [16384, 64]

    // workspace layout: xh (4MB) | xl (4MB) | rowmean (64KB) | s (64KB) | wsum
    __bf16* xh      = (__bf16*)d_ws;
    __bf16* xl      = xh + (size_t)NN * DF;
    float*  rowmean = (float*)(xl + (size_t)NN * DF);
    float*  s       = rowmean + NN;
    float*  wsum    = s + NN;

    split_kernel  <<<(NN * DF) / 256, 256, 0, stream>>>(x, xh, xl);
    rowmean_kernel<<<NN / 8,          256, 0, stream>>>(x, rowmean, s);
    wsum_kernel   <<<1,               64,  0, stream>>>(W, wsum);
    pair_kernel   <<<NN / 64,         256, 0, stream>>>(xh, xl, rowmean, s);
    out_kernel    <<<(NN * HID) / 256,256, 0, stream>>>(s, wsum, b, out);
}
